// MMGCN_51771535786565
// MI455X (gfx1250) — matrix operand order repaired and checked by reading, not compiled, not run on hardware
//
#include <hip/hip_runtime.h>

#define NUM_USER 50000
#define NUM_ITEM 30000
#define NTOT     80000
#define DIMX     64

typedef __attribute__((ext_vector_type(16))) __bf16 bf16x16;
typedef __attribute__((ext_vector_type(8)))  float  floatx8;

__device__ __forceinline__ float leaky01(float x) { return x >= 0.f ? x : 0.01f * x; }

__device__ __forceinline__ unsigned short f32_bf16_rne(float f) {
  union { float f; unsigned u; } c; c.f = f;
  unsigned u = c.u;
  u += 0x7FFFu + ((u >> 16) & 1u);   // round-to-nearest-even
  return (unsigned short)(u >> 16);
}

__device__ __forceinline__ unsigned pack_bf16x2(float a, float b) {
  return (unsigned)f32_bf16_rne(a) | ((unsigned)f32_bf16_rne(b) << 16);
}

// ---------------- elementwise / graph kernels ----------------

__global__ void k_zero(float* __restrict__ p, long long n) {
  long long i = (long long)blockIdx.x * blockDim.x + threadIdx.x;
  if (i < n) p[i] = 0.f;
}

__global__ void k_copy(const float* __restrict__ s, float* __restrict__ d, long long n) {
  long long i = (long long)blockIdx.x * blockDim.x + threadIdx.x;
  if (i < n) d[i] = s[i];
}

__global__ void k_deg_count(const int* __restrict__ dst, float* __restrict__ deg, int E) {
  int i = blockIdx.x * blockDim.x + threadIdx.x;
  if (i < E) atomicAdd(&deg[dst[i]], 1.f);
}

__global__ void k_deg_invert(float* __restrict__ deg, int n) {
  int i = blockIdx.x * blockDim.x + threadIdx.x;
  if (i < n) {
    float d = deg[i];
    deg[i] = (d > 0.f) ? (1.f / fmaxf(d, 1.f)) : 0.f;
  }
}

// one wave (32 lanes) per row; block = 8 waves
__global__ void k_l2norm(float* __restrict__ x, int rows, int d) {
  int w = threadIdx.x >> 5, l = threadIdx.x & 31;
  int r = blockIdx.x * 8 + w;
  if (r >= rows) return;
  float* row = x + (long long)r * d;
  float s = 0.f;
  for (int c = l; c < d; c += 32) { float v = row[c]; s += v * v; }
  #pragma unroll
  for (int m = 16; m > 0; m >>= 1) s += __shfl_xor(s, m, 32);
  float inv = 1.f / fmaxf(sqrtf(s), 1e-12f);
  for (int c = l; c < d; c += 32) row[c] *= inv;
}

// one thread per (edge, channel): coalesced along channels
__global__ void k_scatter(const float* __restrict__ xw, const int* __restrict__ src,
                          const int* __restrict__ dst, float* __restrict__ agg,
                          int d, int E) {
  long long t = (long long)blockIdx.x * blockDim.x + threadIdx.x;
  long long tot = (long long)E * d;
  if (t >= tot) return;
  int e = (int)(t / d), c = (int)(t % d);
  atomicAdd(&agg[(long long)dst[e] * d + c], xw[(long long)src[e] * d + c]);
}

__global__ void k_scale_leaky(float* __restrict__ agg, const float* __restrict__ deg_inv,
                              int rows, int d) {
  long long t = (long long)blockIdx.x * blockDim.x + threadIdx.x;
  long long tot = (long long)rows * d;
  if (t >= tot) return;
  int r = (int)(t / d);
  agg[t] = leaky01(agg[t] * deg_inv[r]);
}

__global__ void k_accum(float* __restrict__ acc, const float* __restrict__ x, long long n) {
  long long i = (long long)blockIdx.x * blockDim.x + threadIdx.x;
  if (i < n) acc[i] += x[i];
}

// rep = sum/3  ->  dot(sum,sum)/9
__global__ void k_score(const float* __restrict__ rep, const int* __restrict__ u,
                        const int* __restrict__ p, const int* __restrict__ ng,
                        float* __restrict__ out, int Bb) {
  int i = blockIdx.x * blockDim.x + threadIdx.x;
  if (i >= Bb) return;
  const float* uv = rep + (long long)u[i]  * DIMX;
  const float* pv = rep + (long long)p[i]  * DIMX;
  const float* nv = rep + (long long)ng[i] * DIMX;
  float sp = 0.f, sn = 0.f;
  #pragma unroll
  for (int c = 0; c < DIMX; ++c) { float uc = uv[c]; sp += uc * pv[c]; sn += uc * nv[c]; }
  out[i]      = sp * (1.f / 9.f);
  out[Bb + i] = sn * (1.f / 9.f);
}

// ---------------- WMMA GEMM (C = act(A@B + bias [+ extra])) ----------------
// modes: 0 = acc, 1 = acc+bias, 2 = leaky(acc+bias)+extra, 3 = leaky(acc+bias+extra)
// Block tile 64x64 (M x N), 8 waves; each wave: one A 16x32 fragment reused by
// two 16x16 WMMA tiles (two f32 accumulators). K chunks of 32.
__global__ __launch_bounds__(256) void k_gemm_wmma(
    const float* __restrict__ A, const float* __restrict__ Bm,
    const float* __restrict__ bias, const float* __restrict__ extra,
    float* __restrict__ C, int M, int Nn, int K, int mode) {
  __shared__ __align__(16) unsigned short As[64][36];   // [m][k], pad keeps 8B/4B alignment
  __shared__ __align__(16) unsigned short Bs[64][36];   // transposed: [n][k]

  int tid  = threadIdx.x;
  int lane = tid & 31, w = tid >> 5;
  int hi = lane >> 4, lo = lane & 15;
  int m0 = blockIdx.x * 64, n0 = blockIdx.y * 64;
  int tm = (w & 3) * 16, tn = (w >> 2) * 32;

  floatx8 acc0 = {0.f,0.f,0.f,0.f,0.f,0.f,0.f,0.f};
  floatx8 acc1 = {0.f,0.f,0.f,0.f,0.f,0.f,0.f,0.f};

  const bool mn_full = (m0 + 64 <= M) && (n0 + 64 <= Nn);

  for (int k0 = 0; k0 < K; k0 += 32) {
    if (mn_full && (k0 + 32 <= K)) {
      // ---- fast path: unpredicated float4 loads, packed bf16 LDS stores ----
      #pragma unroll
      for (int j = 0; j < 2; ++j) {                // A: 64x32 = 512 float4
        int idx4 = tid + 256 * j;
        int r = idx4 >> 3, c = (idx4 & 7) << 2;
        const float4 v = *(const float4*)&A[(size_t)(m0 + r) * K + (k0 + c)];
        uint2 pk; pk.x = pack_bf16x2(v.x, v.y); pk.y = pack_bf16x2(v.z, v.w);
        *(uint2*)&As[r][c] = pk;
        if (j == 0 && k0 + 32 < K)                 // prefetch next K-chunk of A
          __builtin_prefetch(&A[(size_t)(m0 + r) * K + (k0 + 32 + c)], 0, 3);
      }
      #pragma unroll
      for (int j = 0; j < 2; ++j) {                // B: 32x64 = 512 float4
        int idx4 = tid + 256 * j;
        int kk = idx4 >> 4, nn = (idx4 & 15) << 2;
        const float4 v = *(const float4*)&Bm[(size_t)(k0 + kk) * Nn + (n0 + nn)];
        Bs[nn + 0][kk] = f32_bf16_rne(v.x);
        Bs[nn + 1][kk] = f32_bf16_rne(v.y);
        Bs[nn + 2][kk] = f32_bf16_rne(v.z);
        Bs[nn + 3][kk] = f32_bf16_rne(v.w);
      }
    } else {
      // ---- edge path: clamped-address loads + select (branch-free) ----
      #pragma unroll
      for (int j = 0; j < 8; ++j) {                // A: 64x32 scalar
        int i = tid + 256 * j;
        int r = i >> 5, c = i & 31;
        int gr = m0 + r, gc = k0 + c;
        float v = A[(size_t)min(gr, M - 1) * K + min(gc, K - 1)];
        v = (gr < M && gc < K) ? v : 0.f;
        As[r][c] = f32_bf16_rne(v);
      }
      #pragma unroll
      for (int j = 0; j < 8; ++j) {                // B: 32x64 scalar
        int i = tid + 256 * j;
        int kk = i >> 6, nn = i & 63;
        int gk = k0 + kk, gn = n0 + nn;
        float v = Bm[(size_t)min(gk, K - 1) * Nn + min(gn, Nn - 1)];
        v = (gk < K && gn < Nn) ? v : 0.f;
        Bs[nn][kk] = f32_bf16_rne(v);
      }
    }
    __syncthreads();

    // ISA 7.12.2 fragment layouts (wave32):
    // A 16x32: lane holds row M=lo; pair p -> K = (p<4 ? hi*8+2p : 16+hi*8+2(p-4))
    // B 32x16: lane holds col N=lo; pair p -> K = (p<4 ? hi*8+2p : 16+hi*8+2(p-4))
    union { bf16x16 v; unsigned u[8]; } af, bf0, bf1;
    #pragma unroll
    for (int p = 0; p < 8; ++p) {
      int ka = (p < 4) ? (hi * 8 + 2 * p) : (16 + hi * 8 + 2 * (p - 4));
      af.u[p]  = *(const unsigned*)&As[tm + lo][ka];
      int kb = ka;
      bf0.u[p] = *(const unsigned*)&Bs[tn + lo][kb];
      bf1.u[p] = *(const unsigned*)&Bs[tn + 16 + lo][kb];
    }
    acc0 = __builtin_amdgcn_wmma_f32_16x16x32_bf16(
        false, af.v, false, bf0.v, (short)0, acc0, false, false);
    acc1 = __builtin_amdgcn_wmma_f32_16x16x32_bf16(
        false, af.v, false, bf1.v, (short)0, acc1, false, false);
    __syncthreads();
  }

  // epilogue: C row = r + 8*hi, col = lo (16x16 f32 C/D layout), two N-tiles
  #pragma unroll
  for (int r = 0; r < 8; ++r) {
    int row = m0 + tm + r + hi * 8;
    if (row >= M) continue;
    #pragma unroll
    for (int t = 0; t < 2; ++t) {
      int col = n0 + tn + t * 16 + lo;
      if (col >= Nn) continue;
      float v = (t == 0) ? acc0[r] : acc1[r];
      float bv = bias ? bias[col] : 0.f;
      if (mode == 1)      v = v + bv;
      else if (mode == 2) v = leaky01(v + bv) + extra[(size_t)row * Nn + col];
      else if (mode == 3) v = leaky01(v + bv + extra[(size_t)row * Nn + col]);
      C[(size_t)row * Nn + col] = v;
    }
  }
}

// ---------------- host orchestration ----------------

static inline int cdivll(long long a, long long b) { return (int)((a + b - 1) / b); }

extern "C" void kernel_launch(void* const* d_in, const int* in_sizes, int n_in,
                              void* d_out, int out_size, void* d_ws, size_t ws_size,
                              hipStream_t stream) {
  (void)n_in; (void)out_size; (void)ws_size;

  const float* feat_v = (const float*)d_in[0];
  const float* feat_a = (const float*)d_in[1];
  const float* feat_t = (const float*)d_in[2];
  // params (pytree, sorted keys): modalities a,t,v ; 18 sorted leaves each:
  // 0 conv1_w 1 conv2_w 2 conv3_w 3 g1_b 4 g1_w 5 g2_b 6 g2_w 7 g3_b 8 g3_w
  // 9 lin1_b 10 lin1_w 11 lin2_b 12 lin2_w 13 lin3_b 14 lin3_w 15 mlp_b 16 mlp_w 17 pref
  const float* id_emb = (const float*)d_in[57];
  const int* esrc = (const int*)d_in[58];
  const int* edst = (const int*)d_in[59];
  const int* un   = (const int*)d_in[60];
  const int* pn   = (const int*)d_in[61];
  const int* ngn  = (const int*)d_in[62];
  int E  = in_sizes[58];
  int Bb = in_sizes[60];

  float* ws  = (float*)d_ws;
  float* deg = ws;  ws += NTOT;                       // deg -> deg_inv (in place)
  float* rep = ws;  ws += (long long)NTOT * DIMX;     // sum of modality reps
  float* x   = ws;  ws += (long long)NTOT * 128;      // node features per modality
  float* b1  = ws;  ws += (long long)NTOT * 128;      // xw
  float* b2  = ws;  ws += (long long)NTOT * 128;      // agg / h
  float* b3  = ws;  ws += (long long)NTOT * DIMX;     // x_hat

  k_zero<<<cdivll(NTOT, 256), 256, 0, stream>>>(deg, NTOT);
  k_deg_count<<<cdivll(E, 256), 256, 0, stream>>>(edst, deg, E);
  k_deg_invert<<<cdivll(NTOT, 256), 256, 0, stream>>>(deg, NTOT);
  k_zero<<<cdivll((long long)NTOT * DIMX, 256), 256, 0, stream>>>(rep, (long long)NTOT * DIMX);

  struct Mod { const float* feats; int df, dl, pbase; };
  Mod mods[3] = { {feat_a, 128, 128, 3}, {feat_t, 100, 100, 21}, {feat_v, 2048, 128, 39} };

  for (int mi = 0; mi < 3; ++mi) {
    const Mod& md = mods[mi];
    const float* P[18];
    for (int j = 0; j < 18; ++j) P[j] = (const float*)d_in[md.pbase + j];
    const float* conv_w[3] = {P[0], P[1], P[2]};
    const float* g_b[3]    = {P[3], P[5], P[7]};
    const float* g_w[3]    = {P[4], P[6], P[8]};
    const float* lin_b[3]  = {P[9], P[11], P[13]};
    const float* lin_w[3]  = {P[10], P[12], P[14]};
    const float* mlp_b = P[15]; const float* mlp_w = P[16]; const float* pref = P[17];
    int dl = md.dl, df = md.df;

    // x = [pref ; feats@mlp_w + mlp_b]; then row-L2-normalize
    k_copy<<<cdivll((long long)NUM_USER * dl, 256), 256, 0, stream>>>(
        pref, x, (long long)NUM_USER * dl);
    {
      dim3 g(cdivll(NUM_ITEM, 64), cdivll(dl, 64));
      k_gemm_wmma<<<g, 256, 0, stream>>>(md.feats, mlp_w, mlp_b, nullptr,
                                         x + (long long)NUM_USER * dl,
                                         NUM_ITEM, dl, df, /*mode=*/1);
    }
    k_l2norm<<<cdivll(NTOT, 8), 256, 0, stream>>>(x, NTOT, dl);

    int din = dl;
    for (int L = 0; L < 3; ++L) {
      // xw = x @ conv_w
      dim3 ga(cdivll(NTOT, 64), cdivll(din, 64));
      k_gemm_wmma<<<ga, 256, 0, stream>>>(x, conv_w[L], nullptr, nullptr,
                                          b1, NTOT, din, din, /*mode=*/0);
      // agg = deg_inv * scatter_sum(xw[src] -> dst); h = leaky(agg)
      k_zero<<<cdivll((long long)NTOT * din, 256), 256, 0, stream>>>(b2, (long long)NTOT * din);
      k_scatter<<<cdivll((long long)E * din, 256), 256, 0, stream>>>(b1, esrc, edst, b2, din, E);
      k_scale_leaky<<<cdivll((long long)NTOT * din, 256), 256, 0, stream>>>(b2, deg, NTOT, din);
      // x_hat = leaky(x @ lin_w + lin_b) + id_emb
      dim3 gc(cdivll(NTOT, 64), cdivll(DIMX, 64));
      k_gemm_wmma<<<gc, 256, 0, stream>>>(x, lin_w[L], lin_b[L], id_emb,
                                          b3, NTOT, DIMX, din, /*mode=*/2);
      // x = leaky(h @ g_w + g_b + x_hat)
      k_gemm_wmma<<<gc, 256, 0, stream>>>(b2, g_w[L], g_b[L], b3,
                                          x, NTOT, DIMX, din, /*mode=*/3);
      din = DIMX;
    }
    k_accum<<<cdivll((long long)NTOT * DIMX, 256), 256, 0, stream>>>(
        rep, x, (long long)NTOT * DIMX);
  }

  k_score<<<cdivll(Bb, 256), 256, 0, stream>>>(rep, un, pn, ngn, (float*)d_out, Bb);
}
